// RecurrentGraphNeuralNetwork_84636625535580
// MI455X (gfx1250) — matrix operand order repaired and checked by reading, not compiled, not run on hardware
//
#include <hip/hip_runtime.h>
#include <hip/hip_bf16.h>
#include <math.h>

// Problem constants (fixed by the reference)
constexpr int N_NODES = 50000;
constexpr int N_EDGES = 640000;
constexpr int F_IN    = 64;
constexpr int H       = 128;
constexpr int L       = 3;
constexpr int G       = 64;
constexpr int C_OUT   = 2;

typedef __attribute__((ext_vector_type(16))) __bf16 bf16x16;
typedef __attribute__((ext_vector_type(8)))  float  f32x8;

// ---------------------------------------------------------------------------
// Degree / normalization:  dis[n] = rsqrt(deg[n] + 1)
// ---------------------------------------------------------------------------
__global__ void deg_accum_kernel(const int* __restrict__ dst, float* __restrict__ deg) {
    int e = blockIdx.x * blockDim.x + threadIdx.x;
    if (e >= N_EDGES) return;
    atomicAdd(&deg[dst[e]], 1.0f);
}

__global__ void finish_dis_kernel(float* __restrict__ deg) {
    int n = blockIdx.x * blockDim.x + threadIdx.x;
    if (n >= N_NODES) return;
    deg[n] = rsqrtf(deg[n] + 1.0f);   // in-place: deg buffer becomes dis
}

// ---------------------------------------------------------------------------
// WMMA GEMM:  C[M,H] = A[M,K] @ B[H,K]^T (+ bias), K = KSTEPS*32
// One wave32 per 16-row strip; loops over 8 column tiles of 16.
// Fragment layouts per CDNA5 ISA 7.12.2 (bf16 A 16x32, B 32x16, f32 C 16x16).
// ---------------------------------------------------------------------------
template <int KSTEPS>
__global__ __launch_bounds__(256) void wmma_gemm_bias_kernel(
    const float* __restrict__ A, const float* __restrict__ B,
    const float* __restrict__ bias, float* __restrict__ C, int M)
{
    constexpr int K = KSTEPS * 32;
    const int lane = threadIdx.x & 31;
    const int wave = (blockIdx.x * blockDim.x + threadIdx.x) >> 5;
    const int row0 = wave * 16;
    if (row0 >= M) return;                 // uniform per wave (EXEC stays all-1s)

    const int mr = lane & 15;              // A row / B col / C col within tile
    const int kh = lane >> 4;              // K-half selector (0 or 1)

    // Load + convert A fragments for the whole K range (register resident).
    bf16x16 afrag[KSTEPS];
#pragma unroll
    for (int ks = 0; ks < KSTEPS; ++ks) {
        const float* ap = A + (size_t)(row0 + mr) * K + ks * 32 + kh * 8;
#pragma unroll
        for (int i = 0; i < 8; ++i) {
            afrag[ks][i]     = (__bf16)ap[i];        // K = ks*32 + kh*8 + i
            afrag[ks][i + 8] = (__bf16)ap[i + 16];   // K = ks*32 + 16 + kh*8 + i
        }
    }

#pragma unroll
    for (int j0 = 0; j0 < H; j0 += 16) {
        f32x8 acc = {};
#pragma unroll
        for (int ks = 0; ks < KSTEPS; ++ks) {
            // B[k][n] = W[(j0+n)*K + k]; lane n=mr reads K = kh*8+i and 16+kh*8+i.
            const float* bp = B + (size_t)(j0 + mr) * K + ks * 32 + kh * 8;
            bf16x16 bfrag;
#pragma unroll
            for (int i = 0; i < 8; ++i) { bfrag[i] = (__bf16)bp[i]; bfrag[i + 8] = (__bf16)bp[i + 16]; }
            acc = __builtin_amdgcn_wmma_f32_16x16x32_bf16(
                false, afrag[ks], false, bfrag, (short)0, acc, false, false);
        }
        const float bv = bias ? bias[j0 + mr] : 0.0f;
#pragma unroll
        for (int i = 0; i < 8; ++i) {
            C[(size_t)(row0 + i + 8 * kh) * H + (j0 + mr)] = acc[i] + bv;
        }
    }
}

// ---------------------------------------------------------------------------
// GCN aggregation
// ---------------------------------------------------------------------------
__global__ void init_agg_kernel(const float* __restrict__ hw,
                                const float* __restrict__ dis,
                                float* __restrict__ agg) {
    int tid = blockIdx.x * blockDim.x + threadIdx.x;
    if (tid >= N_NODES * H) return;
    int n = tid >> 7;                      // /H
    float dn = dis[n];
    agg[tid] = hw[tid] * dn * dn;          // self-loop contribution
}

// One wave per edge: 32 lanes x 4 floats = 128 features.
__global__ void edge_scatter_kernel(const float* __restrict__ hw,
                                    const int* __restrict__ src,
                                    const int* __restrict__ dst,
                                    const float* __restrict__ dis,
                                    float* __restrict__ agg) {
    long long tid = (long long)blockIdx.x * blockDim.x + threadIdx.x;
    if (tid >= (long long)N_EDGES * 32) return;
    int e = (int)(tid >> 5);
    int q = ((int)tid & 31) << 2;          // feature offset
    int s = src[e], d = dst[e];
    float nrm = dis[s] * dis[d];
    const float4 v = *(const float4*)(hw + (size_t)s * H + q);
    float* out = agg + (size_t)d * H + q;
    atomicAdd(out + 0, v.x * nrm);
    atomicAdd(out + 1, v.y * nrm);
    atomicAdd(out + 2, v.z * nrm);
    atomicAdd(out + 3, v.w * nrm);
}

__global__ void relu_bias_kernel(const float* __restrict__ agg,
                                 const float* __restrict__ bias,
                                 float* __restrict__ h) {
    int tid = blockIdx.x * blockDim.x + threadIdx.x;
    if (tid >= N_NODES * H) return;
    int f = tid & (H - 1);
    h[tid] = fmaxf(agg[tid] + bias[f], 0.0f);
}

// ---------------------------------------------------------------------------
// Global mean pool (atomic accumulate)
// ---------------------------------------------------------------------------
__global__ void pool_accum_kernel(const float* __restrict__ h,
                                  const int* __restrict__ batch,
                                  float* __restrict__ gsum,
                                  float* __restrict__ cnt) {
    int tid = blockIdx.x * blockDim.x + threadIdx.x;
    if (tid >= N_NODES * H) return;
    int n = tid >> 7;
    int f = tid & (H - 1);
    int b = batch[n];
    atomicAdd(&gsum[b * H + f], h[tid]);
    if (f == 0) atomicAdd(&cnt[b], 1.0f);
}

// ---------------------------------------------------------------------------
// Head: mean -> LSTM(seq=1) -> attention(seq=1 => attn==1, ctx=v) -> classifier
// One block per graph, 128 threads, LDS staging.
// ---------------------------------------------------------------------------
__device__ __forceinline__ float sigmoidf(float x) { return 1.0f / (1.0f + expf(-x)); }

__global__ __launch_bounds__(128) void head_kernel(
    const float* __restrict__ gsum, const float* __restrict__ cnt,
    const float* __restrict__ w_ih, const float* __restrict__ b_ih,
    const float* __restrict__ b_hh,
    const float* __restrict__ attn_in_w, const float* __restrict__ attn_in_b,
    const float* __restrict__ attn_out_w, const float* __restrict__ attn_out_b,
    const float* __restrict__ cls_w1, const float* __restrict__ cls_b1,
    const float* __restrict__ cls_w2, const float* __restrict__ cls_b2,
    float* __restrict__ out)
{
    __shared__ float ge[H], hs[H], vv[H], fe[H], zz[H / 2];
    const int g = blockIdx.x;
    const int t = threadIdx.x;

    float cgt = fmaxf(cnt[g], 1.0f);
    ge[t] = gsum[g * H + t] / cgt;
    __syncthreads();

    // LSTM with h0=c0=0: c = sig(i)*tanh(g); hs = sig(o)*tanh(c). (f gate dead.)
    float gi = b_ih[t] + b_hh[t];
    float gg = b_ih[2 * H + t] + b_hh[2 * H + t];
    float go = b_ih[3 * H + t] + b_hh[3 * H + t];
    for (int k = 0; k < H; ++k) {
        float x = ge[k];
        gi += x * w_ih[(size_t)(0 * H + t) * H + k];
        gg += x * w_ih[(size_t)(2 * H + t) * H + k];
        go += x * w_ih[(size_t)(3 * H + t) * H + k];
    }
    float cc = sigmoidf(gi) * tanhf(gg);
    hs[t] = sigmoidf(go) * tanhf(cc);
    __syncthreads();

    // seq_len=1 softmax over one key == 1, so ctx = v (q,k are dead code).
    float acc = attn_in_b[2 * H + t];
    for (int k = 0; k < H; ++k) acc += hs[k] * attn_in_w[(size_t)(2 * H + t) * H + k];
    vv[t] = acc;
    __syncthreads();

    acc = attn_out_b[t];
    for (int k = 0; k < H; ++k) acc += vv[k] * attn_out_w[(size_t)t * H + k];
    fe[t] = acc;
    __syncthreads();

    if (t < H / 2) {
        acc = cls_b1[t];
        for (int k = 0; k < H; ++k) acc += fe[k] * cls_w1[(size_t)t * H + k];
        zz[t] = fmaxf(acc, 0.0f);
    }
    __syncthreads();

    if (t < C_OUT) {
        acc = cls_b2[t];
        for (int k = 0; k < H / 2; ++k) acc += zz[k] * cls_w2[t * (H / 2) + k];
        out[g * C_OUT + t] = acc;
    }
}

// ---------------------------------------------------------------------------
// Launcher
// ---------------------------------------------------------------------------
extern "C" void kernel_launch(void* const* d_in, const int* in_sizes, int n_in,
                              void* d_out, int out_size, void* d_ws, size_t ws_size,
                              hipStream_t stream) {
    const float* x          = (const float*)d_in[0];
    const int*   ei         = (const int*)  d_in[1];
    const int*   batch_idx  = (const int*)  d_in[2];
    const float* emb_w      = (const float*)d_in[3];
    const float* emb_b      = (const float*)d_in[4];
    const float* gcn_w      = (const float*)d_in[5];
    const float* gcn_b      = (const float*)d_in[6];
    const float* lstm_w_ih  = (const float*)d_in[7];
    // d_in[8] = lstm_w_hh (unused: h0 = 0)
    const float* lstm_b_ih  = (const float*)d_in[9];
    const float* lstm_b_hh  = (const float*)d_in[10];
    const float* attn_in_w  = (const float*)d_in[11];
    const float* attn_in_b  = (const float*)d_in[12];
    const float* attn_out_w = (const float*)d_in[13];
    const float* attn_out_b = (const float*)d_in[14];
    const float* cls_w1     = (const float*)d_in[15];
    const float* cls_b1     = (const float*)d_in[16];
    const float* cls_w2     = (const float*)d_in[17];
    const float* cls_b2     = (const float*)d_in[18];
    float* out = (float*)d_out;

    const int* src = ei;
    const int* dst = ei + N_EDGES;

    // Workspace layout (floats)
    float* ws = (float*)d_ws;
    const size_t NH = (size_t)N_NODES * H;
    float* dis  = ws;                      // N (deg, then dis in place)
    float* h    = ws + 50176;              // N*H
    float* hw   = h   + NH;                // N*H
    float* agg  = hw  + NH;                // N*H
    float* gsum = agg + NH;                // G*H
    float* cnt  = gsum + (size_t)G * H;    // G

    const int T = 256;

    // 1) degree -> dis
    hipMemsetAsync(dis, 0, (size_t)N_NODES * sizeof(float), stream);
    hipMemsetAsync(gsum, 0, ((size_t)G * H + G) * sizeof(float), stream);
    deg_accum_kernel<<<(N_EDGES + T - 1) / T, T, 0, stream>>>(dst, dis);
    finish_dis_kernel<<<(N_NODES + T - 1) / T, T, 0, stream>>>(dis);

    // 2) embedding GEMM: h = x @ emb_w.T + emb_b   (K=64 -> 2 wmma K-steps)
    const int waves  = (N_NODES + 15) / 16;          // 3125
    const int gblk   = (waves * 32 + T - 1) / T;     // 391
    wmma_gemm_bias_kernel<2><<<gblk, T, 0, stream>>>(x, emb_w, emb_b, h, N_NODES);

    // 3) GCN layers
    const int eblk  = (int)(((long long)N_EDGES * 32 + T - 1) / T);
    const int nhblk = (int)((NH + T - 1) / T);
    for (int l = 0; l < L; ++l) {
        wmma_gemm_bias_kernel<4><<<gblk, T, 0, stream>>>(
            h, gcn_w + (size_t)l * H * H, nullptr, hw, N_NODES);
        init_agg_kernel<<<nhblk, T, 0, stream>>>(hw, dis, agg);
        edge_scatter_kernel<<<eblk, T, 0, stream>>>(hw, src, dst, dis, agg);
        relu_bias_kernel<<<nhblk, T, 0, stream>>>(agg, gcn_b + (size_t)l * H, h);
    }

    // 4) global mean pool
    pool_accum_kernel<<<nhblk, T, 0, stream>>>(h, batch_idx, gsum, cnt);

    // 5) LSTM + attention + classifier head
    head_kernel<<<G, H, 0, stream>>>(gsum, cnt,
                                     lstm_w_ih, lstm_b_ih, lstm_b_hh,
                                     attn_in_w, attn_in_b, attn_out_w, attn_out_b,
                                     cls_w1, cls_b1, cls_w2, cls_b2, out);
    (void)in_sizes; (void)n_in; (void)out_size; (void)ws_size;
}
